// DisentangledMIPNetwork_88931592831427
// MI455X (gfx1250) — compile-verified, matrix operand order checked by reading, not run on hardware
//
#include <hip/hip_runtime.h>
#include <hip/hip_bf16.h>
#include <math.h>

#define FM       64
#define NV       200000
#define NC       100000
#define NE       2000000
#define EPS_S    1e-6f
#define EPS_NORM 1e-5f

typedef __attribute__((ext_vector_type(16))) _Float16 v16h;
typedef __attribute__((ext_vector_type(8)))  float    v8f;
typedef __attribute__((ext_vector_type(2)))  _Float16 h2;
typedef __attribute__((ext_vector_type(4)))  int      i4;

#if defined(__has_builtin)
#if __has_builtin(__builtin_amdgcn_global_load_async_to_lds_b128) && \
    __has_builtin(__builtin_amdgcn_s_wait_asynccnt)
#define HAVE_ASYNC_LDS 1
#endif
#endif

// GEMM epilogue flags
#define F_INIT   1     // init accumulators from bias (else accumulate from C)
#define F_RELU   2
#define F_SIG    4
#define F_NORM   8     // node_norm + relu (row-wise, in-register)
#define F_OUTER  16    // acc += om[row] * wrow[col]
#define F_ST32   32
#define F_ST16   64
#define F_HEAD   128   // 64->1 dot with wo2, +bo2, sigmoid -> outputs

static __device__ __forceinline__ float half_sum16(float x) {
  #pragma unroll
  for (int m = 8; m; m >>= 1) x += __shfl_xor(x, m, 32);
  return x;
}

// ---------------- elementwise helpers ----------------
__global__ void k_fill(float* __restrict__ p, float v, long n) {
  long t = (long)blockIdx.x * blockDim.x + threadIdx.x;
  if (t < n) p[t] = v;
}
__global__ void k_fill16(_Float16* __restrict__ p, float v, long n) {
  long t = (long)blockIdx.x * blockDim.x + threadIdx.x;
  if (t < n) p[t] = (_Float16)v;
}
__global__ void k_copy(float* __restrict__ d, const float* __restrict__ s, long n) {
  long t = (long)blockIdx.x * blockDim.x + threadIdx.x;
  if (t < n) d[t] = s[t];
}
__global__ void k_conv16(_Float16* __restrict__ d, const float* __restrict__ s, long n) {
  long t = (long)blockIdx.x * blockDim.x + threadIdx.x;
  if (t < n) d[t] = (_Float16)s[t];
}
// WT16[n*ldw + k] = (f16) W[k*ncol + n]   (transpose + convert, padded stride)
__global__ void k_wconvT(_Float16* __restrict__ wt, const float* __restrict__ w,
                         int K, int ncol, int ldw) {
  long t = (long)blockIdx.x * blockDim.x + threadIdx.x;
  if (t >= (long)K * ncol) return;
  int n = (int)(t / K), k = (int)(t % K);
  wt[(long)n * ldw + k] = (_Float16)w[(long)k * ncol + n];
}

// ---------------- sparse path ----------------
__global__ void k_scaler_acc(float* __restrict__ s, const int* __restrict__ idx,
                             const float* __restrict__ vals, int ne) {
  int t = blockIdx.x * blockDim.x + threadIdx.x;
  if (t < ne) atomicAdd(&s[idx[t]], vals[t]);
}
__global__ void k_clamp_eps(float* __restrict__ s, int n) {
  int t = blockIdx.x * blockDim.x + threadIdx.x;
  if (t < n) s[t] = fmaxf(s[t], EPS_S);
}
// Y[sidx[e]] += X[gidx[e]] * val[e] (/ scale[sidx[e]] if scale); one wave per edge
__global__ void k_spmm(const int* __restrict__ gidx, const int* __restrict__ sidx,
                       const float* __restrict__ vals, const float* __restrict__ X,
                       float* __restrict__ Y, const float* __restrict__ scale, int ne) {
  int t = blockIdx.x * blockDim.x + threadIdx.x;
  int e = t >> 5, lane = t & 31;
  if (e >= ne) return;
  int si = sidx[e];
  float v = vals[e];
  if (scale) v /= scale[si];                      // wave-uniform load
  const float2 x = ((const float2*)(X + (size_t)gidx[e] * FM))[lane];
  float* yr = Y + (size_t)si * FM;
  atomicAdd(yr + 2 * lane,     x.x * v);
  atomicAdd(yr + 2 * lane + 1, x.y * v);
}

// ---------------- small fused epilogues ----------------
__global__ void k_loss16(_Float16* __restrict__ l0, _Float16* __restrict__ l1,
                         const float* __restrict__ lhs, const float* __restrict__ cv,
                         const float* __restrict__ cs, long n) {
  long t = (long)blockIdx.x * blockDim.x + threadIdx.x;
  if (t >= n) return;
  long i = t >> 6;
  float d = lhs[t] - cv[i], inv = 1.f / cs[i];
  l0[t] = (_Float16)(fmaxf(d, 0.f)  * inv);
  l1[t] = (_Float16)(fmaxf(-d, 0.f) * inv);
}
__global__ void k_obj16(_Float16* __restrict__ o, const float* __restrict__ q,
                        const float* __restrict__ om, long n) {
  long t = (long)blockIdx.x * blockDim.x + threadIdx.x;
  if (t < n) o[t] = (_Float16)(q[t] * om[t >> 6]);
}

// ------------- WMMA GEMM: C[rows x 64] (+)= A16[rows x 64] @ WT16-chunk -------------
// A: f16 row-major stride 64. WT: f16 column-major (WT[n*ldw + k0 + k]); the 64x64
// chunk is staged through LDS (async-to-LDS when available). One wave per 16x64
// output tile; 4 v8f accumulators x 2 K-steps = 8 v_wmma_f32_16x16x32_f16 per tile.
// Flag-driven fused epilogues (relu/sigmoid/node_norm/rank-1/f16-store/output head).
__global__ __launch_bounds__(256) void k_gemm64(
    const _Float16* __restrict__ A, const _Float16* __restrict__ WT,
    int ldw, int k0, const float* __restrict__ bias,
    float* __restrict__ C, _Float16* __restrict__ C16,
    const float* __restrict__ om, const float* __restrict__ wrow,
    const float* __restrict__ wo2, const float* __restrict__ bo2,
    float* __restrict__ sig_out, float* __restrict__ logit_out,
    int rows, int flags) {
  __shared__ _Float16 wlds[64 * 64];

  // ---- stage the 64x64 f16 weight chunk into LDS (all 256 threads, 16 halves each)
  {
    const int t   = threadIdx.x;
    const int col = t >> 2;
    const int k   = (t & 3) * 16;
    const _Float16* src = WT + (size_t)col * ldw + k0 + k;
    _Float16*       dst = wlds + col * 64 + k;
#if defined(HAVE_ASYNC_LDS)
    __builtin_amdgcn_global_load_async_to_lds_b128(
        (__attribute__((address_space(1))) i4*)src,
        (__attribute__((address_space(3))) i4*)dst, 0, 0);
    __builtin_amdgcn_global_load_async_to_lds_b128(
        (__attribute__((address_space(1))) i4*)(src + 8),
        (__attribute__((address_space(3))) i4*)(dst + 8), 0, 0);
    __builtin_amdgcn_s_wait_asynccnt(0);
#else
    ((uint4*)dst)[0] = ((const uint4*)src)[0];
    ((uint4*)dst)[1] = ((const uint4*)src)[1];
#endif
  }
  __syncthreads();

  const int wave = threadIdx.x >> 5;
  const int lane = threadIdx.x & 31;
  const int row0 = (blockIdx.x * 8 + wave) * 16;
  if (row0 >= rows) return;
  const int r  = lane & 15;
  const int hi = lane >> 4;

  v8f acc[4];
  if (flags & F_INIT) {
    #pragma unroll
    for (int tN = 0; tN < 4; ++tN) {
      float b = bias[tN * 16 + r];
      #pragma unroll
      for (int v = 0; v < 8; ++v) acc[tN][v] = b;
    }
  } else {
    #pragma unroll
    for (int tN = 0; tN < 4; ++tN)
      #pragma unroll
      for (int v = 0; v < 8; ++v)
        acc[tN][v] = C[(size_t)(row0 + hi * 8 + v) * FM + tN * 16 + r];
  }

  const _Float16* arow = A + (size_t)(row0 + r) * FM;

  #pragma unroll
  for (int kk = 0; kk < 64; kk += 32) {
    v16h a;
    #pragma unroll
    for (int v = 0; v < 8; ++v) {
      const int kp = hi * 8 + (v < 4 ? 2 * v : 16 + 2 * (v - 4));
      h2 p = *(const h2*)(arow + kk + kp);
      a[2 * v] = p.x; a[2 * v + 1] = p.y;
    }
    #pragma unroll
    for (int tN = 0; tN < 4; ++tN) {
      const _Float16* wcol = wlds + (tN * 16 + r) * 64 + kk;
      v16h b;
      #pragma unroll
      for (int v = 0; v < 8; ++v) {
        const int kp = hi * 8 + (v < 4 ? 2 * v : 16 + 2 * (v - 4));
        h2 p = *(const h2*)(wcol + kp);
        b[2 * v] = p.x; b[2 * v + 1] = p.y;
      }
      acc[tN] = __builtin_amdgcn_wmma_f32_16x16x32_f16(
          false, a, false, b, (short)0, acc[tN], false, false);
    }
  }

  // ---- fused epilogues ----
  if (flags & F_OUTER) {
    #pragma unroll
    for (int v = 0; v < 8; ++v) {
      float o = om[row0 + hi * 8 + v];
      #pragma unroll
      for (int tN = 0; tN < 4; ++tN) acc[tN][v] += o * wrow[tN * 16 + r];
    }
  }
  if (flags & F_RELU) {
    #pragma unroll
    for (int tN = 0; tN < 4; ++tN)
      #pragma unroll
      for (int v = 0; v < 8; ++v) acc[tN][v] = fmaxf(acc[tN][v], 0.f);
  }
  if (flags & F_SIG) {
    #pragma unroll
    for (int tN = 0; tN < 4; ++tN)
      #pragma unroll
      for (int v = 0; v < 8; ++v) acc[tN][v] = 1.f / (1.f + expf(-acc[tN][v]));
  }
  if (flags & F_NORM) {
    // one row of 64 lives in one 16-lane half across the 4 accumulators
    #pragma unroll
    for (int v = 0; v < 8; ++v) {
      float s  = acc[0][v] + acc[1][v] + acc[2][v] + acc[3][v];
      float ss = acc[0][v] * acc[0][v] + acc[1][v] * acc[1][v] +
                 acc[2][v] * acc[2][v] + acc[3][v] * acc[3][v];
      s  = half_sum16(s);
      ss = half_sum16(ss);
      float m   = s * (1.f / FM);
      float var = ss * (1.f / FM) - m * m;
      float rr  = rsqrtf(var + EPS_NORM);
      #pragma unroll
      for (int tN = 0; tN < 4; ++tN)
        acc[tN][v] = fmaxf((acc[tN][v] - m) * rr, 0.f);
    }
  }
  if (flags & F_HEAD) {
    float w0 = wo2[r], w1 = wo2[16 + r], w2 = wo2[32 + r], w3 = wo2[48 + r];
    float b = bo2[0];
    #pragma unroll
    for (int v = 0; v < 8; ++v) {
      float p = acc[0][v] * w0 + acc[1][v] * w1 + acc[2][v] * w2 + acc[3][v] * w3;
      p = half_sum16(p);
      if (r == 0) {
        int row = row0 + hi * 8 + v;
        float logit = p + b;
        logit_out[row] = logit;
        sig_out[row]   = 1.f / (1.f + expf(-logit));
      }
    }
  }
  if (flags & F_ST32) {
    #pragma unroll
    for (int tN = 0; tN < 4; ++tN)
      #pragma unroll
      for (int v = 0; v < 8; ++v)
        C[(size_t)(row0 + hi * 8 + v) * FM + tN * 16 + r] = acc[tN][v];
  }
  if (flags & F_ST16) {
    #pragma unroll
    for (int tN = 0; tN < 4; ++tN)
      #pragma unroll
      for (int v = 0; v < 8; ++v)
        C16[(size_t)(row0 + hi * 8 + v) * FM + tN * 16 + r] = (_Float16)acc[tN][v];
  }
}

// ------------------------------- host orchestration -------------------------------
extern "C" void kernel_launch(void* const* d_in, const int* in_sizes, int n_in,
                              void* d_out, int out_size, void* d_ws, size_t ws_size,
                              hipStream_t stream) {
  const int*   e_rows = (const int*)  d_in[0];
  const int*   e_cols = (const int*)  d_in[1];
  const float* e_vals = (const float*)d_in[2];
  const float* cvals  = (const float*)d_in[3];
  const float* om     = (const float*)d_in[4];
  // d_in[5] = integer_mask (unused by reference)
  const float* wq1 = (const float*)d_in[6];  const float* bq1 = (const float*)d_in[7];
  const float* wq2 = (const float*)d_in[8];  const float* bq2 = (const float*)d_in[9];
  const float* wc1 = (const float*)d_in[10]; const float* bc1 = (const float*)d_in[11];
  const float* wc2 = (const float*)d_in[12]; const float* bc2 = (const float*)d_in[13];
  const float* wv1 = (const float*)d_in[14]; const float* bv1 = (const float*)d_in[15];
  const float* wv2 = (const float*)d_in[16]; const float* bv2 = (const float*)d_in[17];
  const float* wo1 = (const float*)d_in[18]; const float* bo1 = (const float*)d_in[19];
  const float* wo2 = (const float*)d_in[20]; const float* bo2 = (const float*)d_in[21];
  float* out = (float*)d_out;

  size_t off = 0;
  auto alloc = [&](size_t bytes) -> char* {
    char* p = (char*)d_ws + off;
    off = (off + bytes + 255) & ~(size_t)255;
    return p;
  };
  const long EV = (long)NV * FM, EC = (long)NC * FM;
  float* variables   = (float*)alloc(EV * 4);
  float* constraints = (float*)alloc(EC * 4);
  float* tv    = (float*)alloc(EV * 4);
  float* tc    = (float*)alloc(EC * 4);
  float* query = (float*)alloc(EV * 4);
  float* lhs   = (float*)alloc(EC * 4);
  float* h     = (float*)alloc(EV * 4);
  float* hc    = (float*)alloc(EC * 4);
  float* vsc = (float*)alloc((size_t)NV * 4);
  float* csc = (float*)alloc((size_t)NC * 4);
  _Float16* v16   = (_Float16*)alloc(EV * 2);
  _Float16* tv16  = (_Float16*)alloc(EV * 2);
  _Float16* h16   = (_Float16*)alloc(EV * 2);
  _Float16* obj16 = (_Float16*)alloc(EV * 2);
  _Float16* c16   = (_Float16*)alloc(EC * 2);
  _Float16* tc16  = (_Float16*)alloc(EC * 2);
  _Float16* l016  = (_Float16*)alloc(EC * 2);
  _Float16* l116  = (_Float16*)alloc(EC * 2);
  _Float16* hc16  = (_Float16*)alloc(EC * 2);
  _Float16* wq1T = (_Float16*)alloc((size_t)64 * 128 * 2);
  _Float16* wq2T = (_Float16*)alloc((size_t)64 * 64 * 2);
  _Float16* wc1T = (_Float16*)alloc((size_t)64 * 256 * 2);
  _Float16* wc2T = (_Float16*)alloc((size_t)64 * 64 * 2);
  _Float16* wv1T = (_Float16*)alloc((size_t)64 * 256 * 2);   // 193 padded to ldw=256
  _Float16* wv2T = (_Float16*)alloc((size_t)64 * 64 * 2);
  _Float16* wo1T = (_Float16*)alloc((size_t)64 * 64 * 2);
  (void)ws_size; (void)n_in; (void)in_sizes; (void)out_size;

  const int TPB = 256;
  const int gEV = (int)((EV + TPB - 1) / TPB);
  const int gEC = (int)((EC + TPB - 1) / TPB);
  const int gEdge = (NE * 32 + TPB - 1) / TPB;
  const int gGemmV = (NV / 16 + 7) / 8;
  const int gGemmC = (NC / 16 + 7) / 8;

  auto gemm = [&](int grid, const _Float16* A, const _Float16* WT, int ldw, int k0,
                  const float* bias, float* C, _Float16* C16, int rows, int flags,
                  const float* o = nullptr, const float* wr = nullptr,
                  const float* w2 = nullptr, const float* b2 = nullptr,
                  float* so = nullptr, float* lo = nullptr) {
    k_gemm64<<<grid, TPB, 0, stream>>>(A, WT, ldw, k0, bias, C, C16,
                                       o, wr, w2, b2, so, lo, rows, flags);
  };

  // ---- one-time prep ----
  k_fill<<<gEV, TPB, 0, stream>>>(variables, 1.f, EV);
  k_fill<<<gEC, TPB, 0, stream>>>(constraints, 1.f, EC);
  k_fill16<<<gEV, TPB, 0, stream>>>(v16, 1.f, EV);
  k_fill16<<<gEC, TPB, 0, stream>>>(c16, 1.f, EC);
  k_fill<<<(NV + TPB - 1) / TPB, TPB, 0, stream>>>(vsc, 0.f, NV);
  k_fill<<<(NC + TPB - 1) / TPB, TPB, 0, stream>>>(csc, 0.f, NC);
  k_scaler_acc<<<(NE + TPB - 1) / TPB, TPB, 0, stream>>>(vsc, e_rows, e_vals, NE);
  k_scaler_acc<<<(NE + TPB - 1) / TPB, TPB, 0, stream>>>(csc, e_cols, e_vals, NE);
  k_clamp_eps<<<(NV + TPB - 1) / TPB, TPB, 0, stream>>>(vsc, NV);
  k_clamp_eps<<<(NC + TPB - 1) / TPB, TPB, 0, stream>>>(csc, NC);
  auto wconv = [&](_Float16* wt, const float* w, int K, int ldw) {
    long n = (long)K * 64;
    k_wconvT<<<(int)((n + TPB - 1) / TPB), TPB, 0, stream>>>(wt, w, K, 64, ldw);
  };
  wconv(wq1T, wq1, 128, 128); wconv(wq2T, wq2, 64, 64);
  wconv(wc1T, wc1, 256, 256); wconv(wc2T, wc2, 64, 64);
  wconv(wv1T, wv1, 193, 256); wconv(wv2T, wv2, 64, 64);
  wconv(wo1T, wo1, 64, 64);

  for (int s = 0; s < 3; ++s) {
    // ---- message passing: tc/tv updated in place with scaled atomic scatter ----
    k_copy<<<gEV, TPB, 0, stream>>>(tv, variables, EV);
    k_copy<<<gEC, TPB, 0, stream>>>(tc, constraints, EC);
    for (int p = 0; p < 2; ++p) {
      k_spmm<<<gEdge, TPB, 0, stream>>>(e_rows, e_cols, e_vals, tv, tc, csc, NE);
      k_spmm<<<gEdge, TPB, 0, stream>>>(e_cols, e_rows, e_vals, tc, tv, vsc, NE);
    }
    k_conv16<<<gEV, TPB, 0, stream>>>(tv16, tv, EV);
    k_conv16<<<gEC, TPB, 0, stream>>>(tc16, tc, EC);

    // query = sigmoid(relu([vars|tv]@wq1+bq1)@wq2+bq2)   (relu/f16 & sigmoid fused)
    gemm(gGemmV, v16,  wq1T, 128, 0,  bq1, h, nullptr, NV, F_INIT | F_ST32);
    gemm(gGemmV, tv16, wq1T, 128, 64, nullptr, h, h16, NV, F_RELU | F_ST16);
    gemm(gGemmV, h16,  wq2T, 64, 0, bq2, query, nullptr, NV, F_INIT | F_SIG | F_ST32);

    // lhs = spmm_t(query); constraint losses (f16, pre-scaled)
    k_fill<<<gEC, TPB, 0, stream>>>(lhs, 0.f, EC);
    k_spmm<<<gEdge, TPB, 0, stream>>>(e_rows, e_cols, e_vals, query, lhs, nullptr, NE);
    k_loss16<<<gEC, TPB, 0, stream>>>(l016, l116, lhs, cvals, csc, EC);

    // constraints = relu(norm([c|tc|l0|l1]@wc1+bc1))@wc2+bc2  (norm fused in last chunk)
    gemm(gGemmC, c16,  wc1T, 256, 0,   bc1, hc, nullptr, NC, F_INIT | F_ST32);
    gemm(gGemmC, tc16, wc1T, 256, 64,  nullptr, hc, nullptr, NC, F_ST32);
    gemm(gGemmC, l016, wc1T, 256, 128, nullptr, hc, nullptr, NC, F_ST32);
    gemm(gGemmC, l116, wc1T, 256, 192, nullptr, hc, hc16, NC, F_NORM | F_ST16);
    gemm(gGemmC, hc16, wc2T, 64, 0, bc2, constraints, c16, NC, F_INIT | F_ST32 | F_ST16);

    // variables = relu(norm([vars|tv|q*om|om]@wv1+bv1))@wv2+bv2
    k_obj16<<<gEV, TPB, 0, stream>>>(obj16, query, om, EV);
    gemm(gGemmV, v16,   wv1T, 256, 0,  bv1, h, nullptr, NV, F_INIT | F_ST32);
    gemm(gGemmV, tv16,  wv1T, 256, 64, nullptr, h, nullptr, NV, F_ST32);
    gemm(gGemmV, obj16, wv1T, 256, 128, nullptr, h, h16, NV,
         F_OUTER | F_NORM | F_ST16, om, wv1 + 192 * 64);
    gemm(gGemmV, h16, wv2T, 64, 0, bv2, variables, v16, NV, F_INIT | F_ST32 | F_ST16);

    // output head: relu(norm(vars@wo1+bo1)) . wo2 + bo2 -> sigmoid (fully fused)
    gemm(gGemmV, v16, wo1T, 64, 0, bo1, nullptr, nullptr, NV,
         F_INIT | F_NORM | F_HEAD, nullptr, nullptr, wo2, bo2,
         out + (size_t)s * NV, out + (size_t)3 * NV);
    // 0.2*stop_gradient(x)+0.8*x == x in forward -> no-op
  }
}
